// Performer_7851200217507
// MI455X (gfx1250) — matrix operand order repaired and checked by reading, not compiled, not run on hardware
//
#include <hip/hip_runtime.h>
#include <hip/hip_bf16.h>
#include <math.h>

// Problem constants (from reference setup_inputs): b=8,h=8 -> BH=64
#define BH 64
#define NN 4096
#define DD 64
#define MM 256
#define EE 64
#define KSPLIT 4

typedef __attribute__((ext_vector_type(16))) __bf16 v16bf;
typedef __attribute__((ext_vector_type(8)))  __bf16 v8bf;
typedef __attribute__((ext_vector_type(8)))  float  v8f;

#define WMMA_BF16(A, B, C) \
  __builtin_amdgcn_wmma_f32_16x16x32_bf16(false, (A), false, (B), (short)0, (C), false, false)

#define CONCAT16(LO, HI) \
  __builtin_shufflevector((LO), (HI), 0, 1, 2, 3, 4, 5, 6, 7, 8, 9, 10, 11, 12, 13, 14, 15)

__device__ __forceinline__ __bf16 f2bf(float f) { return (__bf16)f; }   // native v_cvt
__device__ __forceinline__ float bf2f(__bf16 b) { return (float)b; }

// monotonic float<->uint encoding for atomicMax on floats
__device__ __forceinline__ unsigned fenc(float f) {
  unsigned u = __builtin_bit_cast(unsigned, f);
  return (u & 0x80000000u) ? ~u : (u | 0x80000000u);
}
__device__ __forceinline__ float fdec(unsigned e) {
  unsigned u = (e & 0x80000000u) ? (e ^ 0x80000000u) : ~e;
  return __builtin_bit_cast(float, u);
}
__device__ __forceinline__ v8f zero8() {
  v8f c;
  #pragma unroll
  for (int i = 0; i < 8; ++i) c[i] = 0.0f;
  return c;
}

__global__ void init_gmax(unsigned* g) {
  if (threadIdx.x == 0 && blockIdx.x == 0) *g = 0u;  // below any real fenc()
}

// Pbf[m][d] = bf16( d^-0.25 * P[m][d] )  (data_normalizer folded into P)
__global__ void prep_p(const float* __restrict__ P, __bf16* __restrict__ Pbf) {
  int i = blockIdx.x * 256 + threadIdx.x;
  Pbf[i] = f2bf(P[i] * 0.35355339059327379f);
}

// vT[bh][e][n] bf16  <- v[bh][n][e] f32   (64x64 LDS tile transpose)
__global__ void transpose_v(const float* __restrict__ vAll, __bf16* __restrict__ vtAll) {
  __shared__ float tile[64][65];
  const int bh = blockIdx.x, nblk = blockIdx.y, tid = threadIdx.x;
  const float* v = vAll + (size_t)bh * NN * EE + (size_t)nblk * 64 * EE;
  #pragma unroll
  for (int i = 0; i < 16; ++i) {
    int idx = tid + i * 256;
    tile[idx >> 6][idx & 63] = v[idx];
  }
  __syncthreads();
  __bf16* vt = vtAll + (size_t)bh * EE * NN + nblk * 64;
  #pragma unroll
  for (int i = 0; i < 16; ++i) {
    int idx = tid + i * 256;
    int e = idx >> 6, nn = idx & 63;
    vt[(size_t)e * NN + nn] = f2bf(tile[nn][e]);
  }
}

// data_dash = X @ Pbf^T per (b,h); 4 waves, each wave: 16 rows x 256 cols, K=64.
// Finished 16x16 tiles are staged to LDS (one live accumulator -> no spills);
// a running per-row max is kept in registers for the softmax-kernel epilogue.
// is_query: row-max + exp epilogue -> outQ[bh][n][m] bf16
// else:     store (dash - diag) transposed -> outKT[bh][m][n] bf16, atomicMax global max
__global__ __launch_bounds__(128) void feature_gemm(
    const float* __restrict__ X, const __bf16* __restrict__ Pbf,
    __bf16* __restrict__ outQ, __bf16* __restrict__ outKT,
    unsigned* __restrict__ gbits, int is_query) {
  __shared__ float sdash[4][16][256];       // 64 KB of the WGP's 320 KB LDS
  const int bh   = blockIdx.x;
  const int nb   = blockIdx.y;              // 64-row block
  const int w    = threadIdx.x >> 5;        // wave 0..3
  const int lane = threadIdx.x & 31;
  const int half = lane >> 4;               // 0/1
  const int lr   = lane & 15;
  const int n_row = nb * 64 + w * 16 + lr;

  // ---- A fragments (K=0..31 and K=32..63) + sum of squares for diag ----
  const float* xrow = X + ((size_t)bh * NN + n_row) * DD;
  const int kb = half * 8;
  v16bf a0, a1;
  float ss = 0.f;
  #pragma unroll
  for (int v = 0; v < 8; ++v) {
    int K = half * 16 + 2 * v;
    float2 p0 = *(const float2*)(xrow + K);
    float2 p1 = *(const float2*)(xrow + 32 + K);
    ss += p0.x * p0.x + p0.y * p0.y + p1.x * p1.x + p1.y * p1.y;
    a0[2 * v] = f2bf(p0.x); a0[2 * v + 1] = f2bf(p0.y);
    a1[2 * v] = f2bf(p1.x); a1[2 * v + 1] = f2bf(p1.y);
  }
  ss += __shfl_xor(ss, 16, 32);             // lanes l and l+16 cover all 64 K
  const float diag = ss * 0.0625f;          // 0.5 * d^-0.5  (lane holds row lr)

  // ---- 16 column tiles of 16, K-loop of 2 WMMA each; stage tile to LDS ----
  float pm[8];
  #pragma unroll
  for (int v = 0; v < 8; ++v) pm[v] = -3.0e38f;
  for (int ct = 0; ct < 16; ++ct) {
    const __bf16* prow = Pbf + (size_t)(ct * 16 + lr) * DD + half * 16;
    v16bf b0 = *(const v16bf*)(prow);
    v16bf b1 = *(const v16bf*)(prow + 32);
    v8f c = zero8();
    c = WMMA_BF16(a0, b0, c);
    c = WMMA_BF16(a1, b1, c);
    #pragma unroll
    for (int v = 0; v < 8; ++v) {
      pm[v] = fmaxf(pm[v], c[v]);
      sdash[w][v + half * 8][ct * 16 + lr] = c[v];
    }
  }

  float dg[8];
  #pragma unroll
  for (int v = 0; v < 8; ++v) dg[v] = __shfl(diag, v + half * 8, 32);

  if (is_query) {
    #pragma unroll
    for (int v = 0; v < 8; ++v) {           // full row max: reduce 16-lane half
      float m = pm[v];
      m = fmaxf(m, __shfl_xor(m, 1, 32));
      m = fmaxf(m, __shfl_xor(m, 2, 32));
      m = fmaxf(m, __shfl_xor(m, 4, 32));
      m = fmaxf(m, __shfl_xor(m, 8, 32));
      pm[v] = m;
    }
    __bf16* qp = outQ + (size_t)bh * NN * MM;
    for (int ct = 0; ct < 16; ++ct) {
      #pragma unroll
      for (int v = 0; v < 8; ++v) {
        int n = nb * 64 + w * 16 + v + half * 8;
        int m = ct * 16 + lr;
        float dsh = sdash[w][v + half * 8][ct * 16 + lr];
        float val = 0.0625f * (expf(dsh - dg[v] - pm[v]) + 1e-4f);
        qp[(size_t)n * MM + m] = f2bf(val);
      }
    }
  } else {
    float gm = pm[0];
    #pragma unroll
    for (int v = 1; v < 8; ++v) gm = fmaxf(gm, pm[v]);
    gm = fmaxf(gm, __shfl_xor(gm, 1, 32));
    gm = fmaxf(gm, __shfl_xor(gm, 2, 32));
    gm = fmaxf(gm, __shfl_xor(gm, 4, 32));
    gm = fmaxf(gm, __shfl_xor(gm, 8, 32));
    gm = fmaxf(gm, __shfl_xor(gm, 16, 32));
    if (lane == 0) atomicMax(gbits, fenc(gm));
    __bf16* kt = outKT + (size_t)bh * MM * NN;
    for (int ct = 0; ct < 16; ++ct) {
      #pragma unroll
      for (int v = 0; v < 8; ++v) {
        int n = nb * 64 + w * 16 + v + half * 8;
        int m = ct * 16 + lr;
        float dsh = sdash[w][v + half * 8][ct * 16 + lr];
        kt[(size_t)m * NN + n] = f2bf(dsh - dg[v]);   // diag folded now
      }
    }
  }
}

// Fused: k' = ratio*(exp(dash-gmax)+eps) in place on kT, AND row sums -> ksum.
// One wave per row of kT; v8bf vectorized (wave reads/writes 512B per step).
__global__ void k_exp_sum(__bf16* __restrict__ kt,
                          const unsigned* __restrict__ gbits,
                          float* __restrict__ ksum) {
  const int w = threadIdx.x >> 5, lane = threadIdx.x & 31;
  const int row = blockIdx.x * 8 + w;             // bh*MM + m
  const float gmax = fdec(*gbits);
  __bf16* p = kt + (size_t)row * NN;
  float s = 0.f;
  for (int it = 0; it < NN / 256; ++it) {         // 16 iterations
    int base = it * 256 + lane * 8;
    v8bf x = *(const v8bf*)(p + base);
    v8bf y;
    #pragma unroll
    for (int v = 0; v < 8; ++v) {
      float val = 0.0625f * (expf(bf2f(x[v]) - gmax) + 1e-4f);
      y[v] = f2bf(val);
      s += val;
    }
    *(v8bf*)(p + base) = y;
  }
  s += __shfl_xor(s, 1, 32);
  s += __shfl_xor(s, 2, 32);
  s += __shfl_xor(s, 4, 32);
  s += __shfl_xor(s, 8, 32);
  s += __shfl_xor(s, 16, 32);
  if (lane == 0) ksum[row] = s;
}

// Split-K context: partial[z][bh][e][m] = sum over n in z-th quarter of
// kT[m][n]*vT[e][n], via WMMA. Grid (BH, 4 m-groups, KSPLIT). Deterministic:
// each partial slot written exactly once (no atomics).
__global__ void context_gemm(const __bf16* __restrict__ ktAll,
                             const __bf16* __restrict__ vtAll,
                             float* __restrict__ ctxPart) {
  const int bh = blockIdx.x, mg = blockIdx.y, z = blockIdx.z;
  const int w = threadIdx.x >> 5, lane = threadIdx.x & 31;
  const int half = lane >> 4, lr = lane & 15;
  const __bf16* kt = ktAll + (size_t)bh * MM * NN;
  const __bf16* vt = vtAll + (size_t)bh * EE * NN;
  const int m0 = mg * 64 + (w >> 1) * 16;
  const int e0 = (w & 1) * 32;
  const int n_lo = z * (NN / KSPLIT), n_hi = n_lo + NN / KSPLIT;
  const __bf16* arow  = kt + (size_t)(m0 + lr) * NN;
  const __bf16* brow0 = vt + (size_t)(e0 + lr) * NN;
  const __bf16* brow1 = vt + (size_t)(e0 + 16 + lr) * NN;
  const int kb = half * 8, khalf = half * 16;
  v8f c0 = zero8(), c1 = zero8();
  for (int nb = n_lo; nb < n_hi; nb += 32) {
    __builtin_prefetch(arow + nb + 1024, 0, 0);   // global_prefetch_b8
    v8bf lo = *(const v8bf*)(arow + nb + khalf);
    v8bf hi = *(const v8bf*)(arow + nb + khalf + 8);
    v16bf a = CONCAT16(lo, hi);
    v16bf b0 = *(const v16bf*)(brow0 + nb + khalf);
    v16bf b1 = *(const v16bf*)(brow1 + nb + khalf);
    c0 = WMMA_BF16(a, b0, c0);
    c1 = WMMA_BF16(a, b1, c1);
  }
  float* part = ctxPart + ((size_t)z * BH + bh) * EE * MM;   // [e][m] slice
  #pragma unroll
  for (int v = 0; v < 8; ++v) {
    int m = m0 + v + half * 8;
    part[(size_t)(e0 + lr) * MM + m]      = c0[v];
    part[(size_t)(e0 + 16 + lr) * MM + m] = c1[v];
  }
}

// ctxT[bh][e][m] bf16 = sum of KSPLIT f32 partials
__global__ void context_combine(const float* __restrict__ ctxPart,
                                __bf16* __restrict__ ctxT) {
  size_t i = (size_t)blockIdx.x * 256 + threadIdx.x;   // BH*EE*MM total
  const size_t stride = (size_t)BH * EE * MM;
  float s = ctxPart[i] + ctxPart[i + stride] + ctxPart[i + 2 * stride]
          + ctxPart[i + 3 * stride];
  ctxT[i] = f2bf(s);
}

// out[bh][n][e] = (q'[n][:] @ ctx[:,e]) * Dinv[n];  Dinv via LDS reduction.
__global__ void out_gemm(const __bf16* __restrict__ qpAll,
                         const __bf16* __restrict__ ctxTAll,
                         const float* __restrict__ ksumAll,
                         float* __restrict__ outAll) {
  const int bh = blockIdx.x, nblk = blockIdx.y;
  __shared__ float sPart[64][4];
  __shared__ float sDinv[64];
  const __bf16* qp   = qpAll + (size_t)bh * NN * MM;
  const __bf16* ctxT = ctxTAll + (size_t)bh * EE * MM;
  const float* ksum  = ksumAll + bh * MM;
  const int n0 = nblk * 64, tid = threadIdx.x;
  {
    int r = tid >> 2, q4 = tid & 3;
    const __bf16* qrow = qp + (size_t)(n0 + r) * MM + q4 * 64;
    const float* ks = ksum + q4 * 64;
    float s = 0.f;
    #pragma unroll 8
    for (int j = 0; j < 64; ++j) s += bf2f(qrow[j]) * ks[j];
    sPart[r][q4] = s;
  }
  __syncthreads();
  if (tid < 64) {
    float d = sPart[tid][0] + sPart[tid][1] + sPart[tid][2] + sPart[tid][3];
    sDinv[tid] = 1.0f / d;
  }
  __syncthreads();

  const int w = tid >> 5, lane = tid & 31, half = lane >> 4, lr = lane & 15;
  const int rt = w >> 1;             // row tile 0..3
  const int e0 = (w & 1) * 32;       // two e tiles
  const __bf16* arow  = qp + (size_t)(n0 + rt * 16 + lr) * MM;
  const __bf16* brow0 = ctxT + (size_t)(e0 + lr) * MM;
  const __bf16* brow1 = ctxT + (size_t)(e0 + 16 + lr) * MM;
  const int kb = half * 8, khalf = half * 16;
  v8f c0 = zero8(), c1 = zero8();
  #pragma unroll
  for (int kc = 0; kc < MM; kc += 32) {
    v8bf lo = *(const v8bf*)(arow + kc + khalf);
    v8bf hi = *(const v8bf*)(arow + kc + khalf + 8);
    v16bf a = CONCAT16(lo, hi);
    v16bf b0 = *(const v16bf*)(brow0 + kc + khalf);
    v16bf b1 = *(const v16bf*)(brow1 + kc + khalf);
    c0 = WMMA_BF16(a, b0, c0);
    c1 = WMMA_BF16(a, b1, c1);
  }
  float* out = outAll + (size_t)bh * NN * EE;
  #pragma unroll
  for (int v = 0; v < 8; ++v) {
    int r = rt * 16 + v + half * 8;
    float dinv = sDinv[r];
    out[(size_t)(n0 + r) * EE + e0 + lr]      = c0[v] * dinv;
    out[(size_t)(n0 + r) * EE + e0 + 16 + lr] = c1[v] * dinv;
  }
}

extern "C" void kernel_launch(void* const* d_in, const int* in_sizes, int n_in,
                              void* d_out, int out_size, void* d_ws, size_t ws_size,
                              hipStream_t stream) {
  (void)in_sizes; (void)n_in; (void)out_size; (void)ws_size;
  const float* q = (const float*)d_in[0];
  const float* k = (const float*)d_in[1];
  const float* v = (const float*)d_in[2];
  const float* P = (const float*)d_in[3];
  float* out = (float*)d_out;

  char* ws = (char*)d_ws;
  size_t off = 0;
  auto alloc = [&](size_t bytes) -> void* {
    void* p = ws + off;
    off += (bytes + 255) & ~(size_t)255;
    return p;
  };
  __bf16* qp    = (__bf16*)alloc((size_t)BH * NN * MM * 2);        // q'  [bh][n][m]
  __bf16* kt    = (__bf16*)alloc((size_t)BH * MM * NN * 2);        // k'^T [bh][m][n]
  __bf16* vt    = (__bf16*)alloc((size_t)BH * EE * NN * 2);        // v^T [bh][e][n]
  __bf16* ctxT  = (__bf16*)alloc((size_t)BH * EE * MM * 2);        // ctx^T [bh][e][m]
  float* ctxPart = (float*)alloc((size_t)KSPLIT * BH * EE * MM * 4);
  __bf16* Pbf   = (__bf16*)alloc((size_t)MM * DD * 2);
  float*  ksum  = (float*)alloc((size_t)BH * MM * 4);
  unsigned* gb  = (unsigned*)alloc(4);

  init_gmax<<<1, 32, 0, stream>>>(gb);
  prep_p<<<(MM * DD) / 256, 256, 0, stream>>>(P, Pbf);
  transpose_v<<<dim3(BH, NN / 64), 256, 0, stream>>>(v, vt);
  feature_gemm<<<dim3(BH, NN / 64), 128, 0, stream>>>(q, Pbf, qp, nullptr, gb, 1);
  feature_gemm<<<dim3(BH, NN / 64), 128, 0, stream>>>(k, Pbf, nullptr, kt, gb, 0);
  k_exp_sum<<<(BH * MM) / 8, 256, 0, stream>>>(kt, gb, ksum);
  context_gemm<<<dim3(BH, 4, KSPLIT), 256, 0, stream>>>(kt, vt, ctxPart);
  context_combine<<<(BH * EE * MM) / 256, 256, 0, stream>>>(ctxPart, ctxT);
  out_gemm<<<dim3(BH, NN / 64), 256, 0, stream>>>(qp, ctxT, ksum, out);
}
